// GraphAttention_46823733461411
// MI455X (gfx1250) — matrix operand order repaired and checked by reading, not compiled, not run on hardware
//
#include <hip/hip_runtime.h>

// ---------------------------------------------------------------------------
// GAT layer collapsed: softmax(s1[n]+s2[m]+mask) == A[n,m]*exp(s2[m]-c)/rowsum
// => out = relu( (A @ (e⊙WX)) / (A @ e) ), fused over all 8 heads so the
// 268MB adjacency matrix is streamed exactly once through one bf16 WMMA GEMM.
// ---------------------------------------------------------------------------

static constexpr int Nn = 8192;   // nodes
static constexpr int Ff = 512;    // in features
static constexpr int Fo = 64;     // out features per head
static constexpr int Hh = 8;      // heads
static constexpr int NCOLS = 528; // Hh*Fo (512) + 8 denom cols + 8 zero pad = 33 wmma tiles

typedef __bf16 bf16;
typedef __attribute__((ext_vector_type(16))) __bf16 v16bf;
typedef __attribute__((ext_vector_type(8)))  float  v8f;
typedef __attribute__((ext_vector_type(4)))  int    v4i;
// address-space-qualified pointee types for the async-to-LDS builtin
typedef __attribute__((address_space(1))) v4i v4i_g;  // global
typedef __attribute__((address_space(3))) v4i v4i_l;  // LDS

#if defined(__gfx1250__) && __has_builtin(__builtin_amdgcn_global_load_async_to_lds_b128)
#define USE_ASYNC 1
#endif

#if __has_builtin(__builtin_amdgcn_s_wait_asynccnt)
#define WAIT_ASYNC() __builtin_amdgcn_s_wait_asynccnt(0)
#else
#define WAIT_ASYNC() asm volatile("s_wait_asynccnt 0" ::: "memory")
#endif

// Load one 16x32 bf16 WMMA operand fragment from LDS (two 16B chunks per lane,
// chunk addresses follow the CDNA5 ISA 16-bit A/B matrix VGPR layouts).
__device__ __forceinline__ v16bf ld_frag(const bf16* p0, const bf16* p1) {
  union { v16bf v; uint4 q[2]; } u;
  u.q[0] = *(const uint4*)p0;
  u.q[1] = *(const uint4*)p1;
  return u.v;
}

// Monotone float<->uint key for atomicMax-based fp32 max (memset-0 == -inf).
__device__ __forceinline__ unsigned fkey(float f) {
  unsigned b = __float_as_uint(f);
  return (b & 0x80000000u) ? ~b : (b | 0x80000000u);
}
__device__ __forceinline__ float funkey(unsigned k) {
  unsigned b = (k & 0x80000000u) ? (k & 0x7fffffffu) : ~k;
  return __uint_as_float(b);
}

// ---------------------------------------------------------------------------
// Kernel 1: WX[h] = X (8192x512) @ W_h (512x64), bf16 WMMA, f32 accumulate.
// Block = 8 waves; each wave owns 16 rows x 64 cols (4 accum tiles).
// ---------------------------------------------------------------------------
__global__ __launch_bounds__(256) void k_wx(const float* __restrict__ X,
                                            const float* __restrict__ Kw,
                                            bf16* __restrict__ WXbf) {
  __shared__ bf16 XT[128][40];  // 128 rows x 32 k (bf16), +8 pad vs bank conflicts
  __shared__ bf16 WT[64][40];   // col-major: WT[o][k]
  const int h  = blockIdx.y;
  const int m0 = blockIdx.x * 128;
  const int tid = threadIdx.x;
  const int w = tid >> 5, lane = tid & 31, l = lane & 15, hi = lane >> 4;
  const float* W = Kw + (size_t)h * Ff * Fo;
  v8f acc[4] = {};

  for (int k0 = 0; k0 < Ff; k0 += 32) {
    __syncthreads();
    // stage X tile: 128x32 fp32 -> bf16
#pragma unroll
    for (int i = 0; i < 4; ++i) {
      int idx = tid + i * 256;
      int row = idx >> 3, c4 = idx & 7;
      float4 f = *(const float4*)(X + (size_t)(m0 + row) * Ff + k0 + c4 * 4);
      bf16* d = &XT[row][c4 * 4];
      d[0] = (bf16)f.x; d[1] = (bf16)f.y; d[2] = (bf16)f.z; d[3] = (bf16)f.w;
    }
    // stage W tile transposed: WT[o][k] = W[k0+k][o]
#pragma unroll
    for (int i = 0; i < 2; ++i) {
      int idx = tid + i * 256;
      int k = idx >> 4, c4 = idx & 15;
      float4 f = *(const float4*)(W + (size_t)(k0 + k) * Fo + c4 * 4);
      WT[c4 * 4 + 0][k] = (bf16)f.x; WT[c4 * 4 + 1][k] = (bf16)f.y;
      WT[c4 * 4 + 2][k] = (bf16)f.z; WT[c4 * 4 + 3][k] = (bf16)f.w;
    }
    __syncthreads();
    v16bf a = ld_frag(&XT[w * 16 + l][hi * 8], &XT[w * 16 + l][16 + hi * 8]);
#pragma unroll
    for (int j = 0; j < 4; ++j) {
      v16bf b = ld_frag(&WT[j * 16 + l][hi * 8], &WT[j * 16 + l][16 + hi * 8]);
      acc[j] = __builtin_amdgcn_wmma_f32_16x16x32_bf16(false, a, false, b,
                                                       (short)0, acc[j], false, false);
    }
  }
  // C/D layout: lane l = col, VGPR v -> row v + 8*hi
#pragma unroll
  for (int j = 0; j < 4; ++j)
#pragma unroll
    for (int v = 0; v < 8; ++v) {
      int row = m0 + w * 16 + v + 8 * hi;
      WXbf[((size_t)h * Nn + row) * Fo + j * 16 + l] = (bf16)acc[j][v];
    }
}

// ---------------------------------------------------------------------------
// Kernel 2: s2[h,m] = WX[h,m,:] . a2[h]; per-head max via atomic key.
// ---------------------------------------------------------------------------
__global__ __launch_bounds__(256) void k_s2max(const bf16* __restrict__ WXbf,
                                               const float* __restrict__ attnk,
                                               float* __restrict__ s2,
                                               unsigned* __restrict__ maxk) {
  __shared__ float red[256];
  const int h = blockIdx.y;
  const int m = blockIdx.x * 256 + threadIdx.x;
  const bf16* wrow = WXbf + ((size_t)h * Nn + m) * Fo;
  const float* a2 = attnk + h * (2 * Fo) + Fo;
  float acc = 0.f;
#pragma unroll
  for (int o = 0; o < Fo; ++o) acc += (float)wrow[o] * a2[o];
  s2[(size_t)h * Nn + m] = acc;
  red[threadIdx.x] = acc;
  __syncthreads();
  for (int s = 128; s > 0; s >>= 1) {
    if (threadIdx.x < s) red[threadIdx.x] = fmaxf(red[threadIdx.x], red[threadIdx.x + s]);
    __syncthreads();
  }
  if (threadIdx.x == 0) atomicMax(&maxk[h], fkey(red[0]));
}

// ---------------------------------------------------------------------------
// Kernel 3: build V^T (528 x 8192) bf16, K-contiguous for the big GEMM:
//   rows 0..511   : e[h,m] * WX[h,m,o]   (row = h*64+o)
//   rows 512..519 : e[h,m]               (denominator columns)
//   rows 520..527 : 0                    (pad to 33 tiles of 16)
// ---------------------------------------------------------------------------
__global__ __launch_bounds__(256) void k_buildv(const bf16* __restrict__ WXbf,
                                                const float* __restrict__ s2,
                                                const unsigned* __restrict__ maxk,
                                                bf16* __restrict__ Vt) {
  int idx = blockIdx.x * 256 + threadIdx.x;      // 528 * 8192 elements
  int n = idx >> 13, m = idx & (Nn - 1);
  float val = 0.f;
  if (n < Hh * Fo) {
    int h = n >> 6, o = n & 63;
    float e = __expf(s2[(size_t)h * Nn + m] - funkey(maxk[h]));
    val = e * (float)WXbf[((size_t)h * Nn + m) * Fo + o];
  } else if (n < Hh * Fo + Hh) {
    int h = n - Hh * Fo;
    val = __expf(s2[(size_t)h * Nn + m] - funkey(maxk[h]));
  }
  Vt[(size_t)n * Nn + m] = (bf16)val;
}

// ---------------------------------------------------------------------------
// Kernel 4: the money GEMM. Out32(8192x528) = A(8192x8192 fp32 0/1 -> bf16)
// @ V(8192x528 bf16). Block tile 128x176, 8 waves x (16 rows x 11 tiles),
// K-step 32. A converted in-flight; V tile copied LDS-direct (async if avail).
// ---------------------------------------------------------------------------
__global__ __launch_bounds__(256) void k_gemm(const float* __restrict__ A,
                                              const bf16* __restrict__ Vt,
                                              float* __restrict__ Out32) {
  __shared__ bf16 AT[128][40];    // 128 rows x 32 k
  __shared__ bf16 VTl[176][40];   // 176 cols x 32 k (K-contiguous)
  const int m0 = blockIdx.x * 128;
  const int n0 = blockIdx.y * 176;
  const int tid = threadIdx.x;
  const int w = tid >> 5, lane = tid & 31, l = lane & 15, hi = lane >> 4;
  v8f acc[11] = {};

  for (int k0 = 0; k0 < Nn; k0 += 32) {
    __syncthreads();
    // stage A: 128x32 fp32 -> bf16 (in-flight convert)
#pragma unroll
    for (int i = 0; i < 4; ++i) {
      int idx = tid + i * 256;
      int row = idx >> 3, c4 = idx & 7;
      float4 f = *(const float4*)(A + (size_t)(m0 + row) * Nn + k0 + c4 * 4);
      bf16* d = &AT[row][c4 * 4];
      d[0] = (bf16)f.x; d[1] = (bf16)f.y; d[2] = (bf16)f.z; d[3] = (bf16)f.w;
    }
    // stage V^T: 176 rows x 16 k bf16 (k0..k0+15 only), pure 16B copies -> async-to-LDS path
    for (int idx = tid; idx < 352; idx += 256) {
      int n = idx >> 1, c = idx & 1;
      const bf16* src = Vt + (size_t)(n0 + n) * Nn + k0 + c * 8;
      bf16* dst = &VTl[n][c * 8];
#ifdef USE_ASYNC
      __builtin_amdgcn_global_load_async_to_lds_b128((v4i_g*)src, (v4i_l*)dst, 0, 0);
#else
      *(uint4*)dst = *(const uint4*)src;
#endif
    }
#ifdef USE_ASYNC
    WAIT_ASYNC();
#endif
    __syncthreads();
    v16bf a = ld_frag(&AT[w * 16 + l][hi * 8], &AT[w * 16 + l][16 + hi * 8]);
#pragma unroll
    for (int j = 0; j < 11; ++j) {
      v16bf b = ld_frag(&VTl[j * 16 + l][hi * 8], &VTl[j * 16 + l][16 + hi * 8]);
      acc[j] = __builtin_amdgcn_wmma_f32_16x16x32_bf16(false, a, false, b,
                                                       (short)0, acc[j], false, false);
    }
  }
#pragma unroll
  for (int j = 0; j < 11; ++j)
#pragma unroll
    for (int v = 0; v < 8; ++v) {
      int row = m0 + w * 16 + v + 8 * hi;
      Out32[(size_t)row * NCOLS + n0 + j * 16 + l] = acc[j][v];
    }
}

// ---------------------------------------------------------------------------
// Kernel 5: out[n, h*64+o] = relu(numer / denom)
// ---------------------------------------------------------------------------
__global__ __launch_bounds__(256) void k_final(const float* __restrict__ Out32,
                                               float* __restrict__ out) {
  int idx = blockIdx.x * 256 + threadIdx.x;   // 8192*512
  int n = idx >> 9, c = idx & 511;
  int h = c >> 6;
  float num = Out32[(size_t)n * NCOLS + c];
  float den = Out32[(size_t)n * NCOLS + Hh * Fo + h];
  float v = num / den;
  out[idx] = v > 0.f ? v : 0.f;
}

// ---------------------------------------------------------------------------
extern "C" void kernel_launch(void* const* d_in, const int* in_sizes, int n_in,
                              void* d_out, int out_size, void* d_ws, size_t ws_size,
                              hipStream_t stream) {
  (void)in_sizes; (void)n_in; (void)out_size; (void)ws_size;
  const float* X     = (const float*)d_in[0];
  const float* A     = (const float*)d_in[1];
  const float* kern  = (const float*)d_in[2];
  const float* attnk = (const float*)d_in[3];
  float* out = (float*)d_out;

  char* ws = (char*)d_ws;
  bf16*     WXbf  = (bf16*)(ws);                 //  8,388,608 B (8x8192x64 bf16)
  float*    s2    = (float*)(ws + 8388608);      //    262,144 B
  unsigned* maxk  = (unsigned*)(ws + 8650752);   //        256 B
  bf16*     Vt    = (bf16*)(ws + 8651008);       //  8,650,752 B (528x8192 bf16)
  float*    Out32 = (float*)(ws + 17301760);     // 17,301,504 B (8192x528 f32)

  (void)hipMemsetAsync(maxk, 0, 256, stream);                 // key 0 == -inf
  k_wx    <<<dim3(64, 8),  256, 0, stream>>>(X, kern, WXbf);
  k_s2max <<<dim3(32, 8),  256, 0, stream>>>(WXbf, attnk, s2, maxk);
  k_buildv<<<dim3(16896),  256, 0, stream>>>(WXbf, s2, maxk, Vt);
  k_gemm  <<<dim3(64, 3),  256, 0, stream>>>(A, Vt, Out32);
  k_final <<<dim3(16384),  256, 0, stream>>>(Out32, out);
}
